// LinearCrossAttention_65661460021271
// MI455X (gfx1250) — matrix operand order repaired and checked by reading, not compiled, not run on hardware
//
#include <hip/hip_runtime.h>
#include <hip/hip_bf16.h>

// ---------------------------------------------------------------------------
// CDNA5 (gfx1250) cross-attention, bf16 WMMA pipeline with f32 accumulate.
//   q = x @ Wq ; k = ctx @ Wk ; v = ctx @ Wv
//   scores = softmax(q kT * 0.125) ; o = scores v ; out = o @ Wo
// All matmuls use V_WMMA_F32_16X16X32_BF16 (wave32, 16x16 tiles, K=32).
// ---------------------------------------------------------------------------

typedef __bf16 bf16_t;
typedef __attribute__((ext_vector_type(16))) __bf16 v16bf;
typedef __attribute__((ext_vector_type(8)))  __bf16 v8bf;
typedef __attribute__((ext_vector_type(8)))  float  v8f;

#define WMMA_BF16(a, b, c) \
  __builtin_amdgcn_wmma_f32_16x16x32_bf16(false, (a), false, (b), (short)0, (c), false, false)

__device__ __forceinline__ v8f zero8() {
  v8f z = {0.f, 0.f, 0.f, 0.f, 0.f, 0.f, 0.f, 0.f};
  return z;
}

// A-operand (16x32, MxK) loader from row-major bf16 [ld = row stride in elems].
// Lane l holds row m = l&15; kh = l>>4 selects the K sub-block per ISA layout:
//   elems 0..7  -> K = 8*kh + (0..7)      (VGPRs 0-3)
//   elems 8..15 -> K = 16 + 8*kh + (0..7) (VGPRs 4-7)
__device__ __forceinline__ v16bf load_a_g(const bf16_t* __restrict__ base, int ld, int lane) {
  const int m  = lane & 15;
  const int k0 = (lane >> 4) * 8;
  const bf16_t* p = base + (size_t)m * ld + k0;
  v8bf lo = *(const v8bf*)p;
  v8bf hi = *(const v8bf*)(p + 16);
  v16bf a;
#pragma unroll
  for (int i = 0; i < 8; ++i) { a[i] = lo[i]; a[i + 8] = hi[i]; }
  return a;
}

// B-operand (32x16, KxN) loader from TRANSPOSED storage Bt (N x K row-major).
// Lane l holds col n = l&15, K = 8*(l>>4)+(0..7) then 16+8*(l>>4)+(0..7).
__device__ __forceinline__ v16bf load_bt_g(const bf16_t* __restrict__ bt, int ld, int lane) {
  const int n  = lane & 15;
  const int k0 = (lane >> 4) * 8;
  const bf16_t* p = bt + (size_t)n * ld + k0;
  v8bf lo = *(const v8bf*)p;
  v8bf hi = *(const v8bf*)(p + 16);
  v16bf b;
#pragma unroll
  for (int i = 0; i < 8; ++i) { b[i] = lo[i]; b[i + 8] = hi[i]; }
  return b;
}

// ----------------------------- prep kernels --------------------------------
__global__ void cvt_f32_to_bf16(const float* __restrict__ src, bf16_t* __restrict__ dst, int n) {
  int i = blockIdx.x * blockDim.x + threadIdx.x;
  if (i < n) dst[i] = (bf16_t)src[i];
}

// src: R x C f32 row-major  ->  dst: C x R bf16 row-major (transposed)
__global__ void transpose_cvt(const float* __restrict__ src, bf16_t* __restrict__ dst,
                              int R, int C) {
  int i = blockIdx.x * blockDim.x + threadIdx.x;
  if (i < R * C) {
    int r = i / C, c = i - r * C;
    dst[(size_t)c * R + r] = (bf16_t)src[i];
  }
}

// ------------------------- register-blocked GEMM ---------------------------
// C[M,N] = A[M,K] @ Bt[N,K]^T. One wave computes a 32(M) x 64(N) block.
// MODE 0: store bf16 row-major. MODE 1: store f32 row-major (final output).
// MODE 2: V-projection store, transposed per head: vT[((b*8+h)*64+c)*512 + j]
//         where global row = b*512 + j and column = h*64 + c.
template <int MODE>
__global__ __launch_bounds__(32) void gemm_bf16_wmma(const bf16_t* __restrict__ A,
                                                     const bf16_t* __restrict__ Bt,
                                                     void* __restrict__ Cout,
                                                     int M, int N, int K) {
  const int lane = threadIdx.x;
  const int m0 = blockIdx.x * 32;
  const int n0 = blockIdx.y * 64;

  v8f acc[2][4];
#pragma unroll
  for (int mt = 0; mt < 2; ++mt)
#pragma unroll
    for (int nt = 0; nt < 4; ++nt) acc[mt][nt] = zero8();

  for (int k = 0; k < K; k += 32) {
    v16bf a0 = load_a_g(A + (size_t)m0 * K + k, K, lane);
    v16bf a1 = load_a_g(A + (size_t)(m0 + 16) * K + k, K, lane);
#pragma unroll
    for (int nt = 0; nt < 4; ++nt) {
      v16bf b = load_bt_g(Bt + (size_t)(n0 + nt * 16) * K + k, K, lane);
      acc[0][nt] = WMMA_BF16(a0, b, acc[0][nt]);
      acc[1][nt] = WMMA_BF16(a1, b, acc[1][nt]);
    }
  }

  const int n  = lane & 15;
  const int kh = lane >> 4;
#pragma unroll
  for (int mt = 0; mt < 2; ++mt)
#pragma unroll
    for (int nt = 0; nt < 4; ++nt)
#pragma unroll
      for (int r = 0; r < 8; ++r) {
        const int gm  = m0 + mt * 16 + 8 * kh + r;
        const int col = n0 + nt * 16 + n;
        const float val = acc[mt][nt][r];
        if (MODE == 0) {
          ((bf16_t*)Cout)[(size_t)gm * N + col] = (bf16_t)val;
        } else if (MODE == 1) {
          ((float*)Cout)[(size_t)gm * N + col] = val;
        } else {
          const int bb = gm >> 9, j = gm & 511;      // rows are b*512 + j
          const int h = col >> 6, c = col & 63;      // cols are h*64 + c
          ((bf16_t*)Cout)[((size_t)(bb * 8 + h) * 64 + c) * 512 + j] = (bf16_t)val;
        }
      }
}

// ------------------------------ attention ----------------------------------
// grid = (128 i-blocks, 32 b*h). One wave handles 16 query rows x all 512 keys.
// q: [b,2048,512] bf16 ; k: [b,512,512] bf16 ; vT: [b*h, 64, 512] bf16
// aout: [b,2048,512] bf16 (heads re-merged).
__global__ __launch_bounds__(32) void attn_wmma(const bf16_t* __restrict__ q,
                                                const bf16_t* __restrict__ k,
                                                const bf16_t* __restrict__ vT,
                                                bf16_t* __restrict__ aout) {
  __shared__ float sc[16][512];   // f32 scores / exp-scores for this row block
  __shared__ float pmax[32];
  __shared__ float psum[32];
  __shared__ float rinv[16];

  const int lane = threadIdx.x;
  const int ib = blockIdx.x;             // 16-row block of i
  const int bh = blockIdx.y;             // b*8 + h
  const int b = bh >> 3, h = bh & 7;

  const bf16_t* qbase = q + ((size_t)(b * 2048 + ib * 16)) * 512 + h * 64;
  const bf16_t* kbase = k + ((size_t)b * 512) * 512 + h * 64;

  // Q tile A-operands for the two K=32 steps of dim_head=64.
  v16bf qa0 = load_a_g(qbase, 512, lane);
  v16bf qa1 = load_a_g(qbase + 32, 512, lane);

  const int n  = lane & 15;
  const int kh = lane >> 4;

  // scores = (Q Kt) * scale, 32 j-tiles of 16
  for (int jt = 0; jt < 32; ++jt) {
    v8f acc = zero8();
    const bf16_t* kb = kbase + (size_t)jt * 16 * 512;
    v16bf b0 = load_bt_g(kb, 512, lane);
    v16bf b1 = load_bt_g(kb + 32, 512, lane);
    acc = WMMA_BF16(qa0, b0, acc);
    acc = WMMA_BF16(qa1, b1, acc);
#pragma unroll
    for (int r = 0; r < 8; ++r) sc[r + 8 * kh][jt * 16 + n] = acc[r] * 0.125f;
  }
  __syncthreads();

  // softmax: 2 lanes per row (lane, lane+16) each own 256 columns
  const int row = lane & 15;
  const int hv  = lane >> 4;
  {
    const float4* rp = (const float4*)(&sc[row][hv * 256]);
    float mx = -3.0e38f;
#pragma unroll 4
    for (int t = 0; t < 64; ++t) {
      float4 u = rp[t];
      mx = fmaxf(mx, fmaxf(fmaxf(u.x, u.y), fmaxf(u.z, u.w)));
    }
    pmax[lane] = mx;
  }
  __syncthreads();
  {
    float mx = fmaxf(pmax[row], pmax[row + 16]);
    float4* wp = (float4*)(&sc[row][hv * 256]);
    float s = 0.f;
#pragma unroll 4
    for (int t = 0; t < 64; ++t) {
      float4 u = wp[t];
      u.x = __expf(u.x - mx); u.y = __expf(u.y - mx);
      u.z = __expf(u.z - mx); u.w = __expf(u.w - mx);
      s += (u.x + u.y) + (u.z + u.w);
      wp[t] = u;
    }
    psum[lane] = s;
  }
  __syncthreads();
  if (lane < 16) rinv[lane] = 1.0f / (psum[lane] + psum[lane + 16]);
  __syncthreads();

  // o = exp_scores @ V (normalize afterwards; softmax scale is linear in V)
  v8f oacc[4];
#pragma unroll
  for (int nt = 0; nt < 4; ++nt) oacc[nt] = zero8();

  const bf16_t* vbase = vT + (size_t)bh * 64 * 512;
  for (int kt = 0; kt < 16; ++kt) {        // K = 512 keys, step 32
    v16bf a;
    {
      const float* rowp = &sc[lane & 15][kt * 32];
#pragma unroll
      for (int i = 0; i < 8; ++i) {
        a[i]     = (bf16_t)rowp[8 * kh + i];
        a[i + 8] = (bf16_t)rowp[16 + 8 * kh + i];
      }
    }
#pragma unroll
    for (int nt = 0; nt < 4; ++nt) {
      v16bf bv = load_bt_g(vbase + (size_t)(nt * 16) * 512 + kt * 32, 512, lane);
      oacc[nt] = WMMA_BF16(a, bv, oacc[nt]);
    }
  }

  bf16_t* obase = aout + ((size_t)(b * 2048 + ib * 16)) * 512 + h * 64;
#pragma unroll
  for (int nt = 0; nt < 4; ++nt)
#pragma unroll
    for (int r = 0; r < 8; ++r) {
      const int m = r + 8 * kh;
      obase[(size_t)m * 512 + nt * 16 + n] = (bf16_t)(oacc[nt][r] * rinv[m]);
    }
}

// ------------------------------- launcher ----------------------------------
extern "C" void kernel_launch(void* const* d_in, const int* in_sizes, int n_in,
                              void* d_out, int out_size, void* d_ws, size_t ws_size,
                              hipStream_t stream) {
  const float* x   = (const float*)d_in[0];  // (4,2048,1024)
  const float* ctx = (const float*)d_in[1];  // (4,512,768)
  const float* Wq  = (const float*)d_in[2];  // (1024,512)
  const float* Wk  = (const float*)d_in[3];  // (768,512)
  const float* Wv  = (const float*)d_in[4];  // (768,512)
  const float* Wo  = (const float*)d_in[5];  // (512,1024)
  float* out = (float*)d_out;                // (4,2048,1024) f32

  char* ws = (char*)d_ws;
  size_t off = 0;
  auto alloc = [&](size_t bytes) {
    void* p = ws + off;
    off += (bytes + 255) & ~(size_t)255;
    return p;
  };
  bf16_t* xbf  = (bf16_t*)alloc((size_t)8192 * 1024 * 2);  // 16 MB
  bf16_t* cbf  = (bf16_t*)alloc((size_t)2048 * 768 * 2);   //  3 MB
  bf16_t* WqT  = (bf16_t*)alloc((size_t)512 * 1024 * 2);
  bf16_t* WkT  = (bf16_t*)alloc((size_t)512 * 768 * 2);
  bf16_t* WvT  = (bf16_t*)alloc((size_t)512 * 768 * 2);
  bf16_t* WoT  = (bf16_t*)alloc((size_t)1024 * 512 * 2);
  bf16_t* qbf  = (bf16_t*)alloc((size_t)8192 * 512 * 2);   //  8 MB
  bf16_t* kbf  = (bf16_t*)alloc((size_t)2048 * 512 * 2);
  bf16_t* vTbf = (bf16_t*)alloc((size_t)2048 * 512 * 2);   // [b*h,64,512]
  bf16_t* aobf = (bf16_t*)alloc((size_t)8192 * 512 * 2);   //  8 MB
  (void)ws_size; // total ~45 MB

  const int T = 256;
  cvt_f32_to_bf16<<<(8388608 + T - 1) / T, T, 0, stream>>>(x, xbf, 8388608);
  cvt_f32_to_bf16<<<(1572864 + T - 1) / T, T, 0, stream>>>(ctx, cbf, 1572864);
  transpose_cvt<<<(524288 + T - 1) / T, T, 0, stream>>>(Wq, WqT, 1024, 512);
  transpose_cvt<<<(393216 + T - 1) / T, T, 0, stream>>>(Wk, WkT, 768, 512);
  transpose_cvt<<<(393216 + T - 1) / T, T, 0, stream>>>(Wv, WvT, 768, 512);
  transpose_cvt<<<(524288 + T - 1) / T, T, 0, stream>>>(Wo, WoT, 512, 1024);

  // projections
  gemm_bf16_wmma<0><<<dim3(8192 / 32, 512 / 64), 32, 0, stream>>>(xbf, WqT, qbf, 8192, 512, 1024);
  gemm_bf16_wmma<0><<<dim3(2048 / 32, 512 / 64), 32, 0, stream>>>(cbf, WkT, kbf, 2048, 512, 768);
  gemm_bf16_wmma<2><<<dim3(2048 / 32, 512 / 64), 32, 0, stream>>>(cbf, WvT, vTbf, 2048, 512, 768);

  // fused attention (scores + softmax + attn@V)
  attn_wmma<<<dim3(2048 / 16, 32), 32, 0, stream>>>(qbf, kbf, vTbf, aobf);

  // output projection -> f32
  gemm_bf16_wmma<1><<<dim3(8192 / 32, 1024 / 64), 32, 0, stream>>>(aobf, WoT, out, 8192, 1024, 512);
}
